// MultiScaleVectorQuantizerEMA_3281355014169
// MI455X (gfx1250) — matrix operand order repaired and checked by reading, not compiled, not run on hardware
//
#include <hip/hip_runtime.h>
#include <hip/hip_bf16.h>

typedef __attribute__((ext_vector_type(16))) _Float16 v16h;
typedef __attribute__((ext_vector_type(8)))  float    v8f;

#define BATCH 64
#define CH    256
#define NE    8192
#define NPIX  (BATCH*16*16)        // 16384 pixels
#define NTOT  (BATCH*CH*16*16)     // 4194304 elements

// ---------------------------------------------------------------- helpers
__device__ __forceinline__ float cubic_w(float t) {
    const float a = -0.75f;
    t = fabsf(t);
    if (t <= 1.f)  return ((a + 2.f) * t - (a + 3.f)) * t * t + 1.f;
    if (t <  2.f)  return ((a * t - 5.f * a) * t + 8.f * a) * t - 4.f * a;
    return 0.f;
}

// ---------------------------------------------------------------- init kernels
__global__ void k_init(const float* __restrict__ z, float* z_rest, float* z_hat) {
    int i = blockIdx.x * 256 + threadIdx.x;
    z_rest[i] = z[i];
    z_hat[i]  = 0.f;
}

__global__ void k_emb_cvt(const float* __restrict__ emb, _Float16* emb_h) {
    int i = blockIdx.x * 256 + threadIdx.x;
    emb_h[i] = (_Float16)emb[i];
}

__global__ void k_esq(const float* __restrict__ emb, float* esq) {
    int e = blockIdx.x * 256 + threadIdx.x;
    const float* r = emb + (size_t)e * CH;
    float s = 0.f;
    for (int c = 0; c < CH; ++c) { float v = r[c]; s += v * v; }
    esq[e] = s;
}

// conv_w [S,O,I,3,3] fp32  ->  wt [S, rs, O, I] f16  (B-matrix: column o, contiguous k=i)
__global__ void k_wcvt(const float* __restrict__ w, _Float16* wt) {
    int i  = blockIdx.x * 256 + threadIdx.x;     // dest flat: ((si*9+rs)*256+o)*256+ii
    int ii = i & 255;
    int o  = (i >> 8) & 255;
    int t  = i >> 16;                            // si*9+rs
    int rs = t % 9, si = t / 9;
    int r = rs / 3, s = rs % 3;
    wt[i] = (_Float16)w[((((size_t)si * CH + o) * CH + ii) * 3 + r) * 3 + s];
}

__global__ void k_zero_counts(unsigned int* counts) {
    int i = blockIdx.x * 256 + threadIdx.x;
    if (i < NE) counts[i] = 0u;
}

// ---------------------------------------------------------------- downsample (area pool) -> zd f16 [N,256]
__global__ void k_down(const float* __restrict__ z_rest, _Float16* zd, int pn, int last) {
    int c = threadIdx.x;
    int nrow = blockIdx.x;
    float val;
    if (last) {
        int b = nrow >> 8, hw = nrow & 255, h = hw >> 4, w = hw & 15;
        val = z_rest[(((size_t)b * CH + c) * 16 + h) * 16 + w];
    } else {
        int pp = pn * pn;
        int b = nrow / pp, rem = nrow % pp, p = rem / pn, q = rem % pn;
        int sp = (p * 16) / pn, ep = ((p + 1) * 16 + pn - 1) / pn;
        int sq = (q * 16) / pn, eq = ((q + 1) * 16 + pn - 1) / pn;
        const float* base = z_rest + ((size_t)b * CH + c) * 256;
        float s = 0.f;
        for (int h = sp; h < ep; ++h)
            for (int w = sq; w < eq; ++w) s += base[h * 16 + w];
        val = s / (float)((ep - sp) * (eq - sq));
    }
    zd[(size_t)nrow * CH + c] = (_Float16)val;
}

// ---------------------------------------------------------------- WMMA argmin GEMM
// score(row,e) = esq[e] - 2*dot(zd[row], emb[e]); argmin over 8192 codes.
// Two column tiles per iteration (independent WMMA chains) + software-pipelined
// (double-buffered) B fragments so k-step kt+1 loads are in flight during kt's WMMAs.
__global__ void k_argmin(const _Float16* __restrict__ zd, const _Float16* __restrict__ emb_h,
                         const float* __restrict__ esq, int* idx, unsigned int* counts, int N) {
    __shared__ _Float16 lA[16 * CH];
    __shared__ float rv[8 * 16];
    __shared__ int   ri[8 * 16];
    int tid = threadIdx.x, strip = blockIdx.x;

    // stage 16-row A strip in LDS
    const unsigned* src = (const unsigned*)(zd + (size_t)strip * 16 * CH);
    unsigned* dst = (unsigned*)lA;
    for (int t = tid; t < 16 * CH / 2; t += 256) dst[t] = src[t];
    __syncthreads();

    int wave = tid >> 5, lane = tid & 31;
    int hiA = (lane < 16) ? 0 : 16;
    int hiB = (lane < 16) ? 0 : 16;
    const _Float16* arow = lA + (lane & 15) * CH;

    v16h A[8];
#pragma unroll
    for (int kt = 0; kt < 8; ++kt) {
        int k0 = kt * 32;
#pragma unroll
        for (int j = 0; j < 8; ++j) {
            A[kt][j]     = arow[k0 + hiA + j];
            A[kt][8 + j] = arow[k0 + hiA + 8 + j];
        }
    }

    float mv[8]; int mi[8];
#pragma unroll
    for (int q = 0; q < 8; ++q) { mv[q] = 3.4e38f; mi[q] = 0; }

    // tiles {wave*2, wave*2+1} + 16k : 32 iterations x 2 tiles = 64 tiles/wave
    for (int ct = wave * 2; ct < NE / 16; ct += 16) {
        int col0 = ct * 16 + (lane & 15);
        int col1 = col0 + 16;
        const _Float16* cp0 = emb_h + (size_t)col0 * CH;
        const _Float16* cp1 = emb_h + (size_t)col1 * CH;
        // prefetch next iteration's B columns (gfx1250 global_prefetch_b8)
        __builtin_prefetch(cp0 + 16 * 16 * CH, 0, 3);
        __builtin_prefetch(cp1 + 16 * 16 * CH, 0, 3);
        float es0 = esq[col0];
        float es1 = esq[col1];
        v8f acc0, acc1;
#pragma unroll
        for (int q = 0; q < 8; ++q) { acc0[q] = 0.f; acc1[q] = 0.f; }

        // software pipeline: preload kt=0, then load kt+1 before using kt
        v16h B0a, B1a;
#pragma unroll
        for (int j = 0; j < 16; ++j) { B0a[j] = cp0[hiB + j]; B1a[j] = cp1[hiB + j]; }
#pragma unroll
        for (int kt = 0; kt < 8; ++kt) {
            v16h B0n, B1n;
            if (kt < 7) {
                int kn = (kt + 1) * 32 + hiB;
#pragma unroll
                for (int j = 0; j < 16; ++j) { B0n[j] = cp0[kn + j]; B1n[j] = cp1[kn + j]; }
            }
            acc0 = __builtin_amdgcn_wmma_f32_16x16x32_f16(false, A[kt], false, B0a,
                                                          (short)0, acc0, false, false);
            acc1 = __builtin_amdgcn_wmma_f32_16x16x32_f16(false, A[kt], false, B1a,
                                                          (short)0, acc1, false, false);
            if (kt < 7) { B0a = B0n; B1a = B1n; }
        }

#pragma unroll
        for (int q = 0; q < 8; ++q) {
            float sc0 = es0 - 2.f * acc0[q];
            float sc1 = es1 - 2.f * acc1[q];
            if (sc0 < mv[q] || (sc0 == mv[q] && col0 < mi[q])) { mv[q] = sc0; mi[q] = col0; }
            if (sc1 < mv[q] || (sc1 == mv[q] && col1 < mi[q])) { mv[q] = sc1; mi[q] = col1; }
        }
    }

    // reduce across lanes within each 16-lane half (row set matches C layout)
#pragma unroll
    for (int q = 0; q < 8; ++q) {
#pragma unroll
        for (int mask = 1; mask < 16; mask <<= 1) {
            float ov = __shfl_xor(mv[q], mask, 32);
            int   oi = __shfl_xor(mi[q], mask, 32);
            if (ov < mv[q] || (ov == mv[q] && oi < mi[q])) { mv[q] = ov; mi[q] = oi; }
        }
    }
    if ((lane & 15) == 0) {
        int mo = (lane < 16) ? 0 : 8;
#pragma unroll
        for (int q = 0; q < 8; ++q) { rv[wave * 16 + mo + q] = mv[q]; ri[wave * 16 + mo + q] = mi[q]; }
    }
    __syncthreads();
    if (tid < 16) {
        float bv = rv[tid]; int bi = ri[tid];
        for (int w = 1; w < 8; ++w) {
            float ov = rv[w * 16 + tid]; int oi = ri[w * 16 + tid];
            if (ov < bv || (ov == bv && oi < bi)) { bv = ov; bi = oi; }
        }
        int row = strip * 16 + tid;
        if (row < N) { idx[row] = bi; atomicAdd(&counts[bi], 1u); }
    }
}

// ---------------------------------------------------------------- gather + bicubic upsample -> zu (BHWC f32 + f16)
__global__ void k_upsample(const float* __restrict__ emb, const int* __restrict__ idx,
                           float* zu_f, _Float16* zu_h, int pn, int last) {
    int c = threadIdx.x;
    int pix = blockIdx.x;                // b*256 + h*16 + w  (== BHWC pixel index)
    float val;
    if (last) {
        val = emb[(size_t)idx[pix] * CH + c];
    } else {
        int b = pix >> 8, h = (pix >> 4) & 15, w = pix & 15;
        float scale = (float)pn / 16.f;
        float xh = ((float)h + 0.5f) * scale - 0.5f;
        float xw = ((float)w + 0.5f) * scale - 0.5f;
        int fh = (int)floorf(xh), fw = (int)floorf(xw);
        float wh[4], ww[4]; int ph[4], pw[4];
#pragma unroll
        for (int k = 0; k < 4; ++k) {
            int t = fh + k - 1;
            wh[k] = cubic_w(xh - (float)t);
            ph[k] = min(max(t, 0), pn - 1);
            t = fw + k - 1;
            ww[k] = cubic_w(xw - (float)t);
            pw[k] = min(max(t, 0), pn - 1);
        }
        int base = b * pn * pn;
        float s = 0.f;
#pragma unroll
        for (int a = 0; a < 4; ++a) {
            float rowv = 0.f;
#pragma unroll
            for (int e = 0; e < 4; ++e) {
                int id = idx[base + ph[a] * pn + pw[e]];
                rowv += ww[e] * emb[(size_t)id * CH + c];
            }
            s += wh[a] * rowv;
        }
        val = s;
    }
    zu_f[(size_t)pix * CH + c] = val;
    zu_h[(size_t)pix * CH + c] = (_Float16)val;
}

// ---------------------------------------------------------------- WMMA conv3x3 + fused residual update
// strip = (b,h): 16 output pixels x 256 outC. 9 taps of [16,256]x[256,256] f16 GEMM,
// with double-buffered weight (B) fragments to keep loads ahead of the WMMAs.
__global__ void k_conv(const _Float16* __restrict__ zu_h, const float* __restrict__ zu_f,
                       const _Float16* __restrict__ wt, const float* __restrict__ convb,
                       float* z_hat, float* z_rest, int si) {
    __shared__ _Float16 sA[3 * 16 * CH];       // input rows h-1,h,h+1 (BHWC, zero-padded)
    int tid = threadIdx.x, strip = blockIdx.x;
    int b = strip >> 4, h = strip & 15;

    for (int t = tid; t < 3 * 16 * CH / 2; t += 256) {
        int rr  = t / (16 * CH / 2);
        int rem = t % (16 * CH / 2);
        int hs  = h + rr - 1;
        unsigned v = 0u;
        if (hs >= 0 && hs < 16) {
            const unsigned* src = (const unsigned*)(zu_h + (size_t)((b * 16 + hs) * 16) * CH);
            v = src[rem];
        }
        ((unsigned*)sA)[t] = v;
    }
    __syncthreads();

    int wave = tid >> 5, lane = tid & 31;
    int hiA = (lane < 16) ? 0 : 16;
    int hiB = (lane < 16) ? 0 : 16;
    int m = lane & 15;                          // A row = output pixel w

    v8f acc0, acc1;
#pragma unroll
    for (int q = 0; q < 8; ++q) { acc0[q] = 0.f; acc1[q] = 0.f; }
    int ct0 = wave * 2, ct1 = wave * 2 + 1;
    const _Float16* wbase = wt + (size_t)si * 9 * CH * CH;

    for (int rs = 0; rs < 9; ++rs) {
        int r = rs / 3, s = rs % 3;
        int hs = h + r - 1;
        if (hs < 0 || hs > 15) continue;
        int wsrc = m + s - 1;
        bool inb = (wsrc >= 0 && wsrc < 16);
        const _Float16* arow = sA + ((r * 16 + (inb ? wsrc : 0)) * CH);
        const _Float16* cp0 = wbase + (size_t)(rs * CH + ct0 * 16 + (lane & 15)) * CH;
        const _Float16* cp1 = wbase + (size_t)(rs * CH + ct1 * 16 + (lane & 15)) * CH;

        v16h B0a, B1a;
#pragma unroll
        for (int j = 0; j < 16; ++j) { B0a[j] = cp0[hiB + j]; B1a[j] = cp1[hiB + j]; }
#pragma unroll
        for (int kt = 0; kt < 8; ++kt) {
            v16h B0n, B1n;
            if (kt < 7) {
                int kn = (kt + 1) * 32 + hiB;
#pragma unroll
                for (int j = 0; j < 16; ++j) { B0n[j] = cp0[kn + j]; B1n[j] = cp1[kn + j]; }
            }
            int k0 = kt * 32;
            v16h Af;
#pragma unroll
            for (int j = 0; j < 8; ++j) {
                Af[j]     = inb ? arow[k0 + hiA + j]      : (_Float16)0.f;
                Af[8 + j] = inb ? arow[k0 + hiA + 8 + j]  : (_Float16)0.f;
            }
            acc0 = __builtin_amdgcn_wmma_f32_16x16x32_f16(false, Af, false, B0a, (short)0, acc0, false, false);
            acc1 = __builtin_amdgcn_wmma_f32_16x16x32_f16(false, Af, false, B1a, (short)0, acc1, false, false);
            if (kt < 7) { B0a = B0n; B1a = B1n; }
        }
    }

    // fused epilogue: y = conv + bias; res = 0.5*zu + 0.5*y; z_hat += res; z_rest -= res
#pragma unroll
    for (int t = 0; t < 2; ++t) {
        v8f acc = t ? acc1 : acc0;
        int ct  = t ? ct1 : ct0;
        int o   = ct * 16 + (lane & 15);
        float bias = convb[si * CH + o];
#pragma unroll
        for (int q = 0; q < 8; ++q) {
            int mm = q + ((lane < 16) ? 0 : 8);                   // output pixel w
            float y   = acc[q] + bias;
            size_t j  = (size_t)((b * 16 + h) * 16 + mm) * CH + o; // BHWC
            float res = 0.5f * (zu_f[j] + y);
            size_t i  = (size_t)((b * CH + o) * 16 + h) * 16 + mm; // BCHW
            z_hat[i]  += res;
            z_rest[i] -= res;
        }
    }
}

// ---------------------------------------------------------------- deterministic loss reduction
__global__ void k_loss_part(const float* __restrict__ z_hat, const float* __restrict__ z,
                            float* partials) {
    __shared__ float s[256];
    int i = blockIdx.x * 256 + threadIdx.x;
    float d = z_hat[i] - z[i];
    s[threadIdx.x] = d * d;
    __syncthreads();
    for (int o = 128; o > 0; o >>= 1) {
        if (threadIdx.x < o) s[threadIdx.x] += s[threadIdx.x + o];
        __syncthreads();
    }
    if (threadIdx.x == 0) partials[blockIdx.x] = s[0];
}

__global__ void k_loss_red(const float* __restrict__ partials, float* loss_parts, int si) {
    __shared__ float s[256];
    float acc = 0.f;
    for (int t = threadIdx.x; t < 16384; t += 256) acc += partials[t];
    s[threadIdx.x] = acc;
    __syncthreads();
    for (int o = 128; o > 0; o >>= 1) {
        if (threadIdx.x < o) s[threadIdx.x] += s[threadIdx.x + o];
        __syncthreads();
    }
    if (threadIdx.x == 0) loss_parts[si] = 0.25f * s[0] / (float)NTOT;
}

// ---------------------------------------------------------------- final packing: [z_hat | mean_loss | counts]
__global__ void k_final(const float* __restrict__ z_hat, const float* __restrict__ loss_parts,
                        const unsigned int* __restrict__ counts, float* out) {
    int i = blockIdx.x * 256 + threadIdx.x;
    if (i < NTOT) {
        out[i] = z_hat[i];
    } else if (i == NTOT) {
        float s = 0.f;
        for (int k = 0; k < 8; ++k) s += loss_parts[k];
        out[i] = s / 8.f;
    } else if (i < NTOT + 1 + NE) {
        out[i] = (float)counts[i - NTOT - 1];
    }
}

// ---------------------------------------------------------------- launcher
extern "C" void kernel_launch(void* const* d_in, const int* in_sizes, int n_in,
                              void* d_out, int out_size, void* d_ws, size_t ws_size,
                              hipStream_t stream) {
    (void)in_sizes; (void)n_in; (void)out_size; (void)ws_size;
    const float* z     = (const float*)d_in[0];   // [64,256,16,16]
    const float* emb   = (const float*)d_in[1];   // [8192,256]
    const float* convw = (const float*)d_in[2];   // [8,256,256,3,3]
    const float* convb = (const float*)d_in[3];   // [8,256]

    char* w = (char*)d_ws;
    float*     z_rest     = (float*)(w + 0);              // 16 MB
    float*     z_hat      = (float*)(w + 16777216);       // 16 MB
    float*     zu_f       = (float*)(w + 33554432);       // 16 MB
    _Float16*  zd_h       = (_Float16*)(w + 50331648);    // 8 MB
    _Float16*  zu_h       = (_Float16*)(w + 58720256);    // 8 MB
    _Float16*  emb_h      = (_Float16*)(w + 67108864);    // 4 MB
    _Float16*  wt         = (_Float16*)(w + 71303168);    // 9 MB
    float*     esq        = (float*)(w + 80740352);
    int*       idxb       = (int*)(w + 80773120);
    unsigned*  counts     = (unsigned*)(w + 80838656);
    float*     loss_parts = (float*)(w + 80871424);
    float*     partials   = (float*)(w + 80871456);

    k_init<<<NTOT / 256, 256, 0, stream>>>(z, z_rest, z_hat);
    k_emb_cvt<<<NE * CH / 256, 256, 0, stream>>>(emb, emb_h);
    k_esq<<<NE / 256, 256, 0, stream>>>(emb, esq);
    k_wcvt<<<8 * 9 * CH * CH / 256, 256, 0, stream>>>(convw, wt);
    k_zero_counts<<<NE / 256, 256, 0, stream>>>(counts);

    const int pns[8] = {1, 2, 3, 4, 6, 8, 12, 16};
    for (int si = 0; si < 8; ++si) {
        int pn = pns[si];
        int N  = BATCH * pn * pn;               // always a multiple of 16
        int last = (si == 7);
        k_down<<<N, 256, 0, stream>>>(z_rest, zd_h, pn, last);
        k_argmin<<<N / 16, 256, 0, stream>>>(zd_h, emb_h, esq, idxb, counts, N);
        k_upsample<<<NPIX, 256, 0, stream>>>(emb, idxb, zu_f, zu_h, pn, last);
        k_conv<<<BATCH * 16, 256, 0, stream>>>(zu_h, zu_f, wt, convb, z_hat, z_rest, si);
        k_loss_part<<<NTOT / 256, 256, 0, stream>>>(z_hat, z, partials);
        k_loss_red<<<1, 256, 0, stream>>>(partials, loss_parts, si);
    }
    k_final<<<(NTOT + 1 + NE + 255) / 256, 256, 0, stream>>>(z_hat, loss_parts, counts, (float*)d_out);
}
